// MultiHeadAttention_38663295599465
// MI455X (gfx1250) — matrix operand order repaired and checked by reading, not compiled, not run on hardware
//
#include <hip/hip_runtime.h>
#include <hip/hip_bf16.h>

// ---------------------------------------------------------------------------
// MI455X (gfx1250) multi-head attention, wave32 + v_wmma_f32_16x16x32_bf16.
//
// Pipeline (bf16 matmul inputs, fp32 accumulate):
//   1..3) Q/K/V projection GEMMs:  Y = (X @ W^T + b) * scale   (Q: scale=1/8)
//   4)    flash attention per (batch,head): softmax(Q K^T) V
//   5)    output projection GEMM -> fp32 d_out
//
// GEMMs: 128x128 block tile, k-step 64, double-buffered LDS with register
// prefetch (1 barrier / k-step), global_prefetch of the tile 2 steps ahead.
// Attention: double-buffered K/V tiles, online softmax, Q in registers.
//
// Workspace (bf16): Qb | Kb | Vb | attnOut = 4 * 16 MB = 64 MB.
// ---------------------------------------------------------------------------

typedef __attribute__((ext_vector_type(16))) __bf16 v16bf;
typedef __attribute__((ext_vector_type(8)))  __bf16 v8bf;
typedef __attribute__((ext_vector_type(8)))  float  v8f;
typedef __attribute__((ext_vector_type(4)))  float  v4f;

#define EMBED   1024
#define NHEADS  16
#define HDIM    64
#define NBATCH  4
#define SEQ     2048
#define MROWS   (NBATCH * SEQ)   // 8192

// Build a v16bf A/B fragment from two contiguous 16-byte chunks.
__device__ __forceinline__ v16bf pack16(v8bf lo, v8bf hi) {
  v16bf r;
#pragma unroll
  for (int i = 0; i < 8; ++i) { r[i] = lo[i]; r[i + 8] = hi[i]; }
  return r;
}

__device__ __forceinline__ v8f wmma_bf16(v16bf a, v16bf b, v8f c) {
  // (neg_a, A, neg_b, B, c_mod, C, reuse_a, reuse_b)
  return __builtin_amdgcn_wmma_f32_16x16x32_bf16(false, a, false, b,
                                                 (short)0, c, false, false);
}

// ---------------------------------------------------------------------------
// GEMM: Out[M,N] = (A[M,K] @ W[N,K]^T + bias[N]) * scale
//   A: fp32 (converted to bf16 while loading) or bf16; W/bias fp32;
//   Out: bf16 or fp32.  256 threads = 8 waves, wave tile 32x64.
// ---------------------------------------------------------------------------
template <bool A_BF16, bool OUT_F32>
__global__ __launch_bounds__(256) void gemm_bias_kernel(
    const void* __restrict__ Ap, const float* __restrict__ W,
    const float* __restrict__ bias, void* __restrict__ Outp,
    int K, int Nout, float scale)
{
  constexpr int KS  = 64;  // k-step
  constexpr int LDT = 72;  // padded LDS row stride (elements, 144 B)
  __shared__ __align__(16) __bf16 sA[2][128 * LDT];
  __shared__ __align__(16) __bf16 sB[2][128 * LDT];

  const int tid  = threadIdx.x;
  const int lane = tid & 31;
  const int wave = tid >> 5;
  const int wr   = wave >> 1;        // wave row  0..3 (32 rows each)
  const int wc   = wave & 1;         // wave col  0..1 (64 cols each)
  const int hh   = (lane >> 4) & 1;  // lane half
  const int ln   = lane & 15;

  const int rowBase = blockIdx.y * 128;
  const int colBase = blockIdx.x * 128;

  v8f acc[2][4];
#pragma unroll
  for (int i = 0; i < 2; ++i)
#pragma unroll
    for (int f = 0; f < 4; ++f)
#pragma unroll
      for (int r = 0; r < 8; ++r) acc[i][f][r] = 0.0f;

  const int sr = tid >> 1;        // staging row 0..127
  const int sc = (tid & 1) * 32;  // staging col 0/32 (32 elements per thread)

  // --- register-prefetch helpers (convert fp32 -> bf16 at load time) ---
  v8bf ra[4], rb[4];
  auto loadA = [&](int kk) {
    if (A_BF16) {
      const __bf16* A = (const __bf16*)Ap;
      const v8bf* src = (const v8bf*)(A + (size_t)(rowBase + sr) * K + kk + sc);
#pragma unroll
      for (int c = 0; c < 4; ++c) ra[c] = src[c];
    } else {
      const float* A = (const float*)Ap;
      const v4f* src = (const v4f*)(A + (size_t)(rowBase + sr) * K + kk + sc);
#pragma unroll
      for (int c = 0; c < 4; ++c) {
        v4f f0 = src[2 * c], f1 = src[2 * c + 1];
#pragma unroll
        for (int i = 0; i < 4; ++i) {
          ra[c][i]     = (__bf16)f0[i];
          ra[c][i + 4] = (__bf16)f1[i];
        }
      }
    }
  };
  auto loadB = [&](int kk) {
    const v4f* src = (const v4f*)(W + (size_t)(colBase + sr) * K + kk + sc);
#pragma unroll
    for (int c = 0; c < 4; ++c) {
      v4f f0 = src[2 * c], f1 = src[2 * c + 1];
#pragma unroll
      for (int i = 0; i < 4; ++i) {
        rb[c][i]     = (__bf16)f0[i];
        rb[c][i + 4] = (__bf16)f1[i];
      }
    }
  };

  loadA(0);
  loadB(0);

  int p = 0;
  for (int kk = 0; kk < K; kk += KS) {
    // ---- commit the prefetched tile into LDS buffer p ----
#pragma unroll
    for (int c = 0; c < 4; ++c) {
      *(v8bf*)&sA[p][sr * LDT + sc + 8 * c] = ra[c];
      *(v8bf*)&sB[p][sr * LDT + sc + 8 * c] = rb[c];
    }
    __syncthreads();

    // ---- kick off next tile's global loads; L2-prefetch tile after next ----
    if (kk + KS < K) { loadA(kk + KS); loadB(kk + KS); }
    if (kk + 2 * KS < K) {
      if (A_BF16) {
        const __bf16* A = (const __bf16*)Ap;
        __builtin_prefetch(A + (size_t)(rowBase + sr) * K + kk + 2 * KS + sc, 0, 3);
      } else {
        const float* A = (const float*)Ap;
        __builtin_prefetch(A + (size_t)(rowBase + sr) * K + kk + 2 * KS + sc, 0, 3);
      }
      __builtin_prefetch(W + (size_t)(colBase + sr) * K + kk + 2 * KS + sc, 0, 3);
    }

    // ---- 16 WMMAs per wave on buffer p ----
#pragma unroll
    for (int ks = 0; ks < 2; ++ks) {
      v16bf af[2];
#pragma unroll
      for (int i = 0; i < 2; ++i) {
        const int m = wr * 32 + i * 16 + ln;
        const int b = m * LDT + ks * 32;
        af[i] = pack16(*(const v8bf*)&sA[p][b + hh * 16],
                       *(const v8bf*)&sA[p][b + hh * 16 + 8]);
      }
      v16bf bfr[4];
#pragma unroll
      for (int f = 0; f < 4; ++f) {
        const int c = wc * 64 + f * 16 + ln;
        const int b = c * LDT + ks * 32 + hh * 16;
        bfr[f] = pack16(*(const v8bf*)&sB[p][b],
                        *(const v8bf*)&sB[p][b + 8]);
      }
#pragma unroll
      for (int i = 0; i < 2; ++i)
#pragma unroll
        for (int f = 0; f < 4; ++f)
          acc[i][f] = wmma_bf16(af[i], bfr[f], acc[i][f]);
    }
    p ^= 1;
  }

  // ---- epilogue: bias + scale; C element r -> (M = r + 8h, N = ln) ----
  float bv[4];
#pragma unroll
  for (int f = 0; f < 4; ++f) bv[f] = bias[colBase + wc * 64 + f * 16 + ln];

#pragma unroll
  for (int i = 0; i < 2; ++i)
#pragma unroll
    for (int f = 0; f < 4; ++f) {
      const int col = colBase + wc * 64 + f * 16 + ln;
#pragma unroll
      for (int r = 0; r < 8; ++r) {
        const int row = rowBase + wr * 32 + i * 16 + r + 8 * hh;
        const float v = (acc[i][f][r] + bv[f]) * scale;
        if (OUT_F32) ((float*)Outp)[(size_t)row * Nout + col] = v;
        else         ((__bf16*)Outp)[(size_t)row * Nout + col] = (__bf16)v;
      }
    }
}

// ---------------------------------------------------------------------------
// Flash attention: block = (one (n,h) pair, 128 query rows), 8 waves,
// each wave owns 16 q-rows. Key loop in blocks of 64, double-buffered
// K / V^T tiles (1 barrier per block), online softmax, Q in registers.
// ---------------------------------------------------------------------------
__global__ __launch_bounds__(256) void flash_attn_kernel(
    const __bf16* __restrict__ Qb, const __bf16* __restrict__ Kb,
    const __bf16* __restrict__ Vb, __bf16* __restrict__ Ob)
{
  constexpr int LKT = 72;  // padded LDS stride (144 B)
  __shared__ __align__(16) __bf16 sK [2][64 * LKT];     // K block, [s][d]
  __shared__ __align__(16) __bf16 sVT[2][64 * LKT];     // V block transposed [d][s]
  __shared__ __align__(16) __bf16 sP [8 * 16 * LKT];    // per-wave P strip [16][64]

  const int tid  = threadIdx.x;
  const int lane = tid & 31;
  const int wave = tid >> 5;
  const int hh   = (lane >> 4) & 1;
  const int ln   = lane & 15;

  const int n  = blockIdx.x >> 4;   // batch
  const int hd = blockIdx.x & 15;   // head
  const int qb = blockIdx.y * 128;  // query row block
  const size_t base = ((size_t)n * SEQ) * EMBED + (size_t)hd * HDIM;

  // Q A-fragments (16 x 64, two k-steps of 32); Q pre-scaled by 1/8.
  const int qrow = qb + wave * 16;
  v16bf qf[2];
#pragma unroll
  for (int ks = 0; ks < 2; ++ks) {
    const __bf16* qp = Qb + base + (size_t)(qrow + ln) * EMBED + ks * 32;
    qf[ks] = pack16(*(const v8bf*)(qp + hh * 16),
                    *(const v8bf*)(qp + hh * 16 + 8));
  }

  float mrun[8], lrun[8];
  v8f o[4];
#pragma unroll
  for (int r = 0; r < 8; ++r) { mrun[r] = -1e30f; lrun[r] = 0.0f; }
#pragma unroll
  for (int f = 0; f < 4; ++f)
#pragma unroll
    for (int r = 0; r < 8; ++r) o[f][r] = 0.0f;

  const int sr = tid >> 2;        // staging row 0..63
  const int sc = (tid & 3) * 16;  // staging col 0/16/32/48

  // --- register prefetch of one K/V tile ---
  v8bf rk0, rk1, rv0, rv1;
  auto loadKV = [&](int kb) {
    const __bf16* kp = Kb + base + (size_t)(kb + sr) * EMBED + sc;
    rk0 = *(const v8bf*)kp;
    rk1 = *(const v8bf*)(kp + 8);
    const __bf16* vp = Vb + base + (size_t)(kb + sr) * EMBED + sc;
    rv0 = *(const v8bf*)vp;
    rv1 = *(const v8bf*)(vp + 8);
  };
  loadKV(0);

  int p = 0;
  for (int kb = 0; kb < SEQ; kb += 64) {
    // ---- commit prefetched K/V tile into LDS buffer p ----
    *(v8bf*)&sK[p][sr * LKT + sc]     = rk0;
    *(v8bf*)&sK[p][sr * LKT + sc + 8] = rk1;
#pragma unroll
    for (int j = 0; j < 8; ++j) {     // transpose V into sVT[d][s]
      sVT[p][(sc + j)     * LKT + sr] = rv0[j];
      sVT[p][(sc + 8 + j) * LKT + sr] = rv1[j];
    }
    __syncthreads();
    if (kb + 64 < SEQ) loadKV(kb + 64);  // overlap next loads with compute

    // ---- S = Q @ K^T (16 x 64): 4 C-fragments, 2 k-steps each ----
    v8f s[4];
#pragma unroll
    for (int f = 0; f < 4; ++f) {
      const int krow = f * 16 + ln;  // B lane = key row
      v16bf b0 = pack16(*(const v8bf*)&sK[p][krow * LKT + hh * 16],
                        *(const v8bf*)&sK[p][krow * LKT + hh * 16 + 8]);
      v16bf b1 = pack16(*(const v8bf*)&sK[p][krow * LKT + 32 + hh * 16],
                        *(const v8bf*)&sK[p][krow * LKT + 32 + hh * 16 + 8]);
      v8f z;
#pragma unroll
      for (int r = 0; r < 8; ++r) z[r] = 0.0f;
      z    = wmma_bf16(qf[0], b0, z);
      s[f] = wmma_bf16(qf[1], b1, z);
    }

    // ---- online softmax over this 64-key block ----
    __bf16* pw = &sP[wave * 16 * LKT];
#pragma unroll
    for (int r = 0; r < 8; ++r) {
      float mx = s[0][r];
#pragma unroll
      for (int f = 1; f < 4; ++f) mx = fmaxf(mx, s[f][r]);
#pragma unroll
      for (int msk = 8; msk >= 1; msk >>= 1)
        mx = fmaxf(mx, __shfl_xor(mx, msk, 32));   // reduce over 16 N-lanes
      const float mnew  = fmaxf(mrun[r], mx);
      const float alpha = __expf(mrun[r] - mnew);
#pragma unroll
      for (int f = 0; f < 4; ++f) o[f][r] *= alpha;
      float rs = 0.0f;
#pragma unroll
      for (int f = 0; f < 4; ++f) {
        const float pv = __expf(s[f][r] - mnew);
        rs += pv;
        pw[(r + 8 * hh) * LKT + f * 16 + ln] = (__bf16)pv;  // C -> LDS
      }
#pragma unroll
      for (int msk = 8; msk >= 1; msk >>= 1) rs += __shfl_xor(rs, msk, 32);
      lrun[r] = lrun[r] * alpha + rs;
      mrun[r] = mnew;
    }

    // ---- P A-fragments from the per-wave LDS strip ----
    v16bf pf[2];
#pragma unroll
    for (int ks = 0; ks < 2; ++ks) {
      const __bf16* pp = pw + ln * LKT + ks * 32;
      pf[ks] = pack16(*(const v8bf*)(pp + hh * 16),
                      *(const v8bf*)(pp + hh * 16 + 8));
    }
    // ---- O += P @ V : B lane = output dim d, contiguous k = s ----
#pragma unroll
    for (int f = 0; f < 4; ++f) {
      const int drow = f * 16 + ln;
      v16bf b0 = pack16(*(const v8bf*)&sVT[p][drow * LKT + hh * 16],
                        *(const v8bf*)&sVT[p][drow * LKT + hh * 16 + 8]);
      v16bf b1 = pack16(*(const v8bf*)&sVT[p][drow * LKT + 32 + hh * 16],
                        *(const v8bf*)&sVT[p][drow * LKT + 32 + hh * 16 + 8]);
      o[f] = wmma_bf16(pf[0], b0, o[f]);
      o[f] = wmma_bf16(pf[1], b1, o[f]);
    }
    p ^= 1;
  }

  // ---- finalize: O / l, store bf16 into [n, s, h*64+d] layout ----
#pragma unroll
  for (int f = 0; f < 4; ++f)
#pragma unroll
    for (int r = 0; r < 8; ++r) {
      const int row = qrow + r + 8 * hh;
      const float v = o[f][r] / lrun[r];
      Ob[base + (size_t)row * EMBED + f * 16 + ln] = (__bf16)v;
    }
}

// ---------------------------------------------------------------------------
extern "C" void kernel_launch(void* const* d_in, const int* in_sizes, int n_in,
                              void* d_out, int out_size, void* d_ws, size_t ws_size,
                              hipStream_t stream)
{
  const float* q  = (const float*)d_in[0];
  const float* k  = (const float*)d_in[1];
  const float* v  = (const float*)d_in[2];
  const float* wq = (const float*)d_in[3];
  const float* bq = (const float*)d_in[4];
  const float* wk = (const float*)d_in[5];
  const float* bk = (const float*)d_in[6];
  const float* wv = (const float*)d_in[7];
  const float* bv = (const float*)d_in[8];
  const float* wo = (const float*)d_in[9];
  const float* bo = (const float*)d_in[10];

  __bf16* Qb = (__bf16*)d_ws;
  __bf16* Kb = Qb + (size_t)MROWS * EMBED;
  __bf16* Vb = Kb + (size_t)MROWS * EMBED;
  __bf16* Ab = Vb + (size_t)MROWS * EMBED;   // attention output (bf16)

  const dim3 g(EMBED / 128, MROWS / 128);    // (8, 64)
  const float qscale = 0.125f;               // 1/sqrt(HDIM)

  // Q/K/V projections (fold 1/sqrt(d) into Q).
  gemm_bias_kernel<false, false><<<g, 256, 0, stream>>>(q, wq, bq, Qb, EMBED, EMBED, qscale);
  gemm_bias_kernel<false, false><<<g, 256, 0, stream>>>(k, wk, bk, Kb, EMBED, EMBED, 1.0f);
  gemm_bias_kernel<false, false><<<g, 256, 0, stream>>>(v, wv, bv, Vb, EMBED, EMBED, 1.0f);

  // Flash attention: grid = (n*h, S/128).
  flash_attn_kernel<<<dim3(NBATCH * NHEADS, SEQ / 128), 256, 0, stream>>>(Qb, Kb, Vb, Ab);

  // Output projection -> fp32 d_out.
  gemm_bias_kernel<true, true><<<g, 256, 0, stream>>>(Ab, wo, bo, (float*)d_out, EMBED, EMBED, 1.0f);

  (void)in_sizes; (void)n_in; (void)out_size; (void)ws_size;
}
